// ResidualVQ_1331439861820
// MI455X (gfx1250) — matrix operand order repaired and checked by reading, not compiled, not run on hardware
//
#include <hip/hip_runtime.h>
#include <hip/hip_bf16.h>

// ---------------------------------------------------------------------------
// Residual VQ for MI455X (gfx1250): bf16 WMMA cross-term GEMM + fp32 residual,
// with double-buffered GLOBAL_LOAD_ASYNC_TO_LDS codebook streaming (ASYNCcnt).
//   inputs:    [32, 256, 32, 32] f32   -> x[n, c],  n = b*1024 + h*32 + w
//   codebooks: [8, 1024, 256]    f32
//   out = x - final_residual  (== sum of quantized stages == STE value)
// ---------------------------------------------------------------------------

typedef __bf16 bf16_t;
typedef __attribute__((ext_vector_type(16))) __bf16 v16bf;
typedef __attribute__((ext_vector_type(8)))  __bf16 v8bf;
typedef __attribute__((ext_vector_type(8)))  float  v8f;

#define QSTAGES 8
#define KCODES  1024
#define CDIM    256
#define NVECS   32768       // 32 * 32 * 32
#define TN      64          // residual rows per block
#define TK      128         // codebook columns per chunk
#define NCHUNK  (KCODES / TK)
#define LDB     264         // bf16 row stride (pad) for res_bf16 / cb tiles
#define LDR     257         // f32 row stride (pad) for residual
#define SLD     132         // f32 row stride (pad) for scores

// LDS carve-up (bytes, 16B aligned)
#define OFF_RES   0                         // 64*257*4  = 65792
#define OFF_RESB  65792                     // 64*264*2  = 33792
#define OFF_CBA   99584                     // 128*264*2 = 67584
#define OFF_CBB   167168                    // 128*264*2 = 67584
#define OFF_SC    234752                    // 64*132*4  = 33792
#define OFF_BV    268544                    // 256*4
#define OFF_BI    269568                    // 256*4
#define OFF_KW    270592                    // 64*4
#define SMEM_BYTES 270848

__device__ __forceinline__ bf16_t to_bf16(float f) {
    union { float f; unsigned u; } v; v.f = f;
    unsigned r = v.u + 0x7FFFu + ((v.u >> 16) & 1u);   // round-to-nearest-even
    union { unsigned short s; __bf16 b; } o;
    o.s = (unsigned short)(r >> 16);
    return o.b;
}

// Async copy 16B/lane: global -> LDS, tracked by ASYNCcnt (no VGPR staging).
__device__ __forceinline__ void async_b128(unsigned lds_addr, const void* gptr) {
    asm volatile("global_load_async_to_lds_b128 %0, %1, off"
                 :: "v"(lds_addr), "v"(gptr) : "memory");
}

__device__ __forceinline__ void wait_async_16() {
#if __has_builtin(__builtin_amdgcn_s_wait_asynccnt)
    __builtin_amdgcn_s_wait_asynccnt(16);
#else
    asm volatile("s_wait_asynccnt 0x10" ::: "memory");
#endif
}

__device__ __forceinline__ void wait_async_0() {
#if __has_builtin(__builtin_amdgcn_s_wait_asynccnt)
    __builtin_amdgcn_s_wait_asynccnt(0);
#else
    asm volatile("s_wait_asynccnt 0x0" ::: "memory");
#endif
}

// ---------------------------------------------------------------------------
// Prep: codebooks f32 -> bf16 (ws) and halfnorm[q*K+k] = 0.5*sum_c cb^2
// One 256-thread block per codebook row (8192 blocks).
// ---------------------------------------------------------------------------
__global__ void rvq_prep(const float* __restrict__ cb,
                         bf16_t* __restrict__ cbb,
                         float* __restrict__ hn) {
    const int rowb = blockIdx.x;                 // q*1024 + k
    const int t = threadIdx.x;                   // 0..255
    const float v = cb[(size_t)rowb * CDIM + t];
    cbb[(size_t)rowb * CDIM + t] = to_bf16(v);

    float p = v * v;
    #pragma unroll
    for (int off = 16; off > 0; off >>= 1)
        p += __shfl_xor(p, off, 32);
    __shared__ float wsum[8];
    if ((t & 31) == 0) wsum[t >> 5] = p;
    __syncthreads();
    if (t == 0) {
        float s = 0.f;
        #pragma unroll
        for (int i = 0; i < 8; ++i) s += wsum[i];
        hn[rowb] = 0.5f * s;
    }
}

// ---------------------------------------------------------------------------
// Main kernel: 512 blocks x 256 threads (8 wave32), TN=64 rows per block.
// ---------------------------------------------------------------------------
__global__ void rvq_main(const float* __restrict__ x,
                         const float* __restrict__ cb_f32,
                         const bf16_t* __restrict__ cbb,
                         const float* __restrict__ halfnorm,
                         float* __restrict__ out) {
    extern __shared__ unsigned char smem[];
    float*  s_res   = (float*) (smem + OFF_RES);    // [TN][LDR]  fp32 residual
    bf16_t* s_resb  = (bf16_t*)(smem + OFF_RESB);   // [TN][LDB]  bf16 residual
    bf16_t* s_cbA   = (bf16_t*)(smem + OFF_CBA);    // [TK][LDB]  cb buffer A
    bf16_t* s_cbB   = (bf16_t*)(smem + OFF_CBB);    // [TK][LDB]  cb buffer B
    float*  s_sc    = (float*) (smem + OFF_SC);     // [TN][SLD]  scores
    float*  s_bv    = (float*) (smem + OFF_BV);     // [256] partial best val
    int*    s_bi    = (int*)   (smem + OFF_BI);     // [256] partial best idx
    int*    s_kwin  = (int*)   (smem + OFF_KW);     // [TN]  winning code

    const int tid  = threadIdx.x;
    const int lane = tid & 31;
    const int wave = tid >> 5;                      // 0..7 == ntile
    const int n0   = blockIdx.x * TN;               // first row of tile
    const int bimg = n0 >> 10;                      // batch index
    const int hw0  = n0 & 1023;                     // hw offset within batch
    const size_t gbase = (size_t)bimg * CDIM * 1024 + hw0;

    // per-thread async-chunk addressing (16 x 16B per thread per chunk)
    const int cid0  = tid;                          // chunk id stride 256
    // ---- issue very first codebook chunk (stage 0, kc0 = 0) into buffer A --
    {
        const bf16_t* src = cbb;
        #pragma unroll
        for (int i = 0; i < 16; ++i) {
            const int cid = i * 256 + cid0;
            const int row = cid >> 5;
            const int co  = (cid & 31) << 3;
            async_b128((unsigned)(size_t)(s_cbA + row * LDB + co),
                       src + ((size_t)cid << 3));
        }
    }

    // ---- load x tile: s_res (fp32) + s_resb (bf16). Coalesced over rows. ----
    {
        const int row = tid & 63;
        const int c0  = tid >> 6;                   // 0..3
        #pragma unroll 4
        for (int i = 0; i < 64; ++i) {
            const int c = c0 + i * 4;
            const float v = x[gbase + (size_t)c * 1024 + row];
            s_res[row * LDR + c]  = v;
            s_resb[row * LDB + c] = to_bf16(v);
        }
    }
    __syncthreads();

    // ---- per-lane constant WMMA addressing ----
    const int arow_lo = lane & 15;                  // row within M-tile
    const int aoff    = (lane < 16) ? 0 : 16;       // A k sub-offset
    const int boff    = (lane < 16) ? 0 : 16;       // B k sub-offset
    const int colt    = lane & 15;                  // C/D column within tile
    const int rhalf   = (lane >> 4) << 3;           // C/D row half (0 or 8)

    int cur = 0;                                    // buffer holding current chunk

    for (int q = 0; q < QSTAGES; ++q) {
        const float* hnq = halfnorm + q * KCODES;

        float best  = -3.402823466e+38f;
        int   bestk = 0x7fffffff;

        for (int ci = 0; ci < NCHUNK; ++ci) {
            const int kc0 = ci * TK;

            // -- software pipeline: issue next chunk into the other buffer
            const bool is_last = (q == QSTAGES - 1) && (ci == NCHUNK - 1);
            if (!is_last) {
                const int qn  = (ci == NCHUNK - 1) ? q + 1 : q;
                const int cin = (ci == NCHUNK - 1) ? 0 : ci + 1;
                const bf16_t* src =
                    cbb + ((size_t)qn * KCODES + (size_t)cin * TK) * CDIM;
                bf16_t* dst = cur ? s_cbA : s_cbB;
                #pragma unroll
                for (int i = 0; i < 16; ++i) {
                    const int cid = i * 256 + cid0;
                    const int row = cid >> 5;
                    const int co  = (cid & 31) << 3;
                    async_b128((unsigned)(size_t)(dst + row * LDB + co),
                               src + ((size_t)cid << 3));
                }
                wait_async_16();                     // current chunk landed
            } else {
                wait_async_0();
            }
            __syncthreads();

            // -- WMMA: scores[0:64, 0:128] = res_bf16 * cb_chunk^T (f32 acc)
            {
                const bf16_t* s_cb = cur ? s_cbB : s_cbA;
                const int ntile = wave;              // 0..7
                #pragma unroll
                for (int mtile = 0; mtile < 4; ++mtile) {
                    v8f acc = {};
                    const bf16_t* arow = s_resb + (mtile * 16 + arow_lo) * LDB;
                    const bf16_t* brow = s_cb   + (ntile * 16 + colt)    * LDB;
                    #pragma unroll
                    for (int kb = 0; kb < CDIM; kb += 32) {
                        v8bf alo = *(const v8bf*)(arow + kb + aoff);
                        v8bf ahi = *(const v8bf*)(arow + kb + aoff + 8);
                        v16bf a;
                        #pragma unroll
                        for (int i = 0; i < 8; ++i) { a[i] = alo[i]; a[i + 8] = ahi[i]; }
                        v16bf b = *(const v16bf*)(brow + kb + boff);
                        acc = __builtin_amdgcn_wmma_f32_16x16x32_bf16(
                                  false, a, false, b, (short)0, acc, false, false);
                    }
                    const int rbase = mtile * 16 + rhalf;
                    #pragma unroll
                    for (int j = 0; j < 8; ++j)
                        s_sc[(rbase + j) * SLD + ntile * 16 + colt] = acc[j];
                }
            }
            __syncthreads();

            // -- partial argmax: thread = (row, 32-col segment)
            {
                const int arow = tid & 63;
                const int aseg = tid >> 6;           // 0..3
                #pragma unroll 4
                for (int cc = 0; cc < 32; ++cc) {
                    const int col = aseg * 32 + cc;
                    const int k   = kc0 + col;
                    const float s = s_sc[arow * SLD + col] - hnq[k];
                    if (s > best || (s == best && k < bestk)) { best = s; bestk = k; }
                }
            }
            cur ^= 1;
        }

        // -- combine 4 partials per row, first-min tie-break (ascending k cmp)
        s_bv[tid] = best;
        s_bi[tid] = bestk;
        __syncthreads();
        if (tid < TN) {
            float bv = s_bv[tid]; int bk = s_bi[tid];
            #pragma unroll
            for (int s = 1; s < 4; ++s) {
                const float v = s_bv[s * 64 + tid];
                const int   k = s_bi[s * 64 + tid];
                if (v > bv || (v == bv && k < bk)) { bv = v; bk = k; }
            }
            s_kwin[tid] = bk;
        }
        __syncthreads();

        // -- residual update with exact fp32 codebook rows (coalesced gather)
        {
            const float* cbrow = cb_f32 + (size_t)q * KCODES * CDIM;
            for (int r = 0; r < TN; ++r) {
                const int k = s_kwin[r];
                const float cv = cbrow[(size_t)k * CDIM + tid];
                const float nr = s_res[r * LDR + tid] - cv;
                s_res[r * LDR + tid]  = nr;
                s_resb[r * LDB + tid] = to_bf16(nr);
            }
        }
        __syncthreads();
    }

    // ---- epilogue: out = x - final_residual  (== quantized sum / STE value)
    {
        const int row = tid & 63;
        const int c0  = tid >> 6;
        #pragma unroll 4
        for (int i = 0; i < 64; ++i) {
            const int c = c0 + i * 4;
            const size_t g = gbase + (size_t)c * 1024 + row;
            out[g] = x[g] - s_res[row * LDR + c];
        }
    }
}

// ---------------------------------------------------------------------------
extern "C" void kernel_launch(void* const* d_in, const int* in_sizes, int n_in,
                              void* d_out, int out_size, void* d_ws, size_t ws_size,
                              hipStream_t stream) {
    (void)in_sizes; (void)n_in; (void)out_size; (void)ws_size;

    const float* x  = (const float*)d_in[0];   // [32,256,32,32]
    const float* cb = (const float*)d_in[1];   // [8,1024,256]
    float* out = (float*)d_out;

    bf16_t* cbb = (bf16_t*)d_ws;                                   // 4 MB
    float*  hn  = (float*)((char*)d_ws +
                           (size_t)QSTAGES * KCODES * CDIM * sizeof(bf16_t));

    rvq_prep<<<QSTAGES * KCODES, 256, 0, stream>>>(cb, cbb, hn);

    hipFuncSetAttribute((const void*)rvq_main,
                        hipFuncAttributeMaxDynamicSharedMemorySize, SMEM_BYTES);
    rvq_main<<<NVECS / TN, 256, SMEM_BYTES, stream>>>(x, cb, cbb, hn, out);
}
